// WaveNetEncoder_49821620633870
// MI455X (gfx1250) — matrix operand order repaired and checked by reading, not compiled, not run on hardware
//
#include <hip/hip_runtime.h>

typedef __attribute__((ext_vector_type(2)))  float        v2f;
typedef __attribute__((ext_vector_type(4)))  float        v4f;
typedef __attribute__((ext_vector_type(8)))  float        v8f;
typedef __attribute__((ext_vector_type(8)))  _Float16     v8h;
typedef __attribute__((ext_vector_type(16))) _Float16     v16h;
typedef __attribute__((ext_vector_type(4)))  unsigned int v4u;
typedef __attribute__((ext_vector_type(4)))  int          v4i;
typedef __attribute__((ext_vector_type(8)))  int          v8i;

#define LAYERS 10
#define NL     40
#define RC     32
#define DC     32
#define SC     256
#define OUTC   256
#define BATCH  4
#define TLEN   32768
#define BT     (BATCH*TLEN)   /* 131072 */
#define KTOT   (NL*DC)        /* 1280   */

// ---------------------------------------------------------------------------
// LDS byte-offset of a __shared__ object (generic -> AS(3) -> int)
// ---------------------------------------------------------------------------
typedef __attribute__((address_space(3))) void lds_void_t;
__device__ __forceinline__ unsigned lds_off(const void* p) {
  return (unsigned)(unsigned long long)(lds_void_t*)p;
}

// ---------------------------------------------------------------------------
// TDM: DMA a 2D f16 tile (td0 elems x td1 rows, row stride `stride` elems)
// from global memory into LDS at byte offset `loff`.  D# per CDNA5 ISA §8.
// ---------------------------------------------------------------------------
__device__ __forceinline__ void tdm_load_tile(const void* gsrc, unsigned loff,
                                              unsigned td0, unsigned td1,
                                              unsigned stride) {
  unsigned long long ga = (unsigned long long)gsrc;
  v4u g0;
  g0[0] = 1u;                                           // count=1, user D#
  g0[1] = loff;                                         // lds_addr (bytes)
  g0[2] = (unsigned)ga;                                 // global_addr[31:0]
  g0[3] = (unsigned)((ga >> 32) & 0x01ffffffu) | (2u << 30);  // addr[56:32], type=2
  v8i g1;
  g1[0] = 1 << 16;                                      // data_size=1 -> 2 bytes
  g1[1] = (int)(td0 << 16);                             // tensor_dim0[15:0] @bits63:48
  g1[2] = (int)((td0 >> 16) | (td1 << 16));             // dim0 hi | tensor_dim1 lo
  g1[3] = (int)((td1 >> 16) | (td0 << 16));             // dim1 hi | tile_dim0
  g1[4] = (int)(td1 & 0xffffu);                         // tile_dim1 (tile_dim2=0)
  g1[5] = (int)stride;                                  // tensor_dim0_stride lo32
  g1[6] = 0;
  g1[7] = 0;
  v4i z4 = {0, 0, 0, 0};
#if defined(__clang_major__) && (__clang_major__ >= 23)
  v8i z8 = {0, 0, 0, 0, 0, 0, 0, 0};
  __builtin_amdgcn_tensor_load_to_lds(g0, g1, z4, z4, z8, 0);
#else
  __builtin_amdgcn_tensor_load_to_lds(g0, g1, z4, z4, 0);
#endif
}

// ---------------------------------------------------------------------------
// Kernel 0: f32 -> f16 weight conversion (one-shot prologue)
// ---------------------------------------------------------------------------
__global__ __launch_bounds__(256) void k_cvt(const float* __restrict__ src,
                                             _Float16* __restrict__ dst, int n) {
  const int i = blockIdx.x * 256 + threadIdx.x;
  if (i < n) dst[i] = (_Float16)src[i];
}

// ---------------------------------------------------------------------------
// Kernel 1: start 1x1 conv  h0[bt][c] = W_start[c]*x[bt] + b_start[c]
// ---------------------------------------------------------------------------
__global__ __launch_bounds__(256) void k_start(const float* __restrict__ x,
                                               const float* __restrict__ Wst,
                                               const float* __restrict__ bst,
                                               float* __restrict__ h0) {
  const int bt = blockIdx.x * blockDim.x + threadIdx.x;
  const float xv = x[bt];
  float* o = h0 + (size_t)bt * RC;
#pragma unroll
  for (int c = 0; c < RC; c += 4) {
    v4f w = *(const v4f*)(Wst + c);
    v4f b = *(const v4f*)(bst + c);
    v4f r = w * xv + b;
    *(v4f*)(o + c) = r;
  }
}

// ---------------------------------------------------------------------------
// Kernel 2: one residual layer (dilated conv -> tanh -> residual 1x1)
//   f32 WMMA 16x16x4 keeps the 40-layer residual chain at reference precision.
//   h layout: [bt][32] f32.  Also writes f16 row into H_cat [bt][1280].
// ---------------------------------------------------------------------------
__global__ __launch_bounds__(256) void k_layer(const float* __restrict__ hin,
                                               float* __restrict__ hout,
                                               _Float16* __restrict__ hcat,
                                               const float* __restrict__ Wd,
                                               const float* __restrict__ bd,
                                               const float* __restrict__ Wr,
                                               const float* __restrict__ br,
                                               int layer, int dil) {
  __shared__ float zbuf[8][16][36];   // per-wave transposed z tile (padded)

  const int tid  = threadIdx.x;
  const int wv   = tid >> 5;
  const int lane = tid & 31;
  const int l15  = lane & 15;
  const int lh   = lane >> 4;  // 0/1
  const int hi8  = lh * 8;     // C-layout row offset for upper lane half

  // ---- preload per-lane weight A-fragments (held across the 4 time tiles)
  const float* Wd_l = Wd + (size_t)layer * DC * RC * 2;  // [o][c][tap]
  const float* Wr_l = Wr + (size_t)layer * RC * DC;      // [o][c]
  v2f aWs[2][8], aWc[2][8], aWr[2][8];
#pragma unroll
  for (int mt = 0; mt < 2; ++mt) {
    const int M = mt * 16 + l15;
#pragma unroll
    for (int kc = 0; kc < 8; ++kc) {
      const int K = kc * 4 + lh * 2;
      v4f wd = *(const v4f*)(Wd_l + (M * RC + K) * 2);  // {Wd0[K],Wd1[K],Wd0[K+1],Wd1[K+1]}
      v2f s = {wd.x, wd.z};  // tap0 -> shifted input
      v2f c = {wd.y, wd.w};  // tap1 -> current input
      aWs[mt][kc] = s;
      aWc[mt][kc] = c;
      aWr[mt][kc] = *(const v2f*)(Wr_l + M * DC + K);
    }
  }
  // per-C-element biases
  float bdv[2][8], brv[2][8];
#pragma unroll
  for (int mt = 0; mt < 2; ++mt)
#pragma unroll
    for (int r = 0; r < 8; ++r) {
      const int row = mt * 16 + hi8 + r;
      bdv[mt][r] = bd[layer * DC + row];
      brv[mt][r] = br[layer * RC + row];
    }

  const int colBase = blockIdx.x * 512 + wv * 64;  // 8 waves x 64 cols
  for (int nt = 0; nt < 4; ++nt) {
    const int bt = colBase + nt * 16 + l15;  // this lane's time column
    const int t  = bt & (TLEN - 1);          // time within its batch row

    // ---- z = Wd1 @ h[t] + Wd0 @ h[t-d]
    v8f accz[2] = {};
#pragma unroll
    for (int kc = 0; kc < 8; ++kc) {
      const int K = kc * 4 + lh * 2;
      v2f bh  = *(const v2f*)(hin + (size_t)bt * RC + K);
      v2f bhs = {0.0f, 0.0f};
      if (t >= dil) bhs = *(const v2f*)(hin + (size_t)(bt - dil) * RC + K);
#pragma unroll
      for (int mt = 0; mt < 2; ++mt) {
        accz[mt] = __builtin_amdgcn_wmma_f32_16x16x4_f32(
            false, aWc[mt][kc], false, bh, (short)0, accz[mt], false, false);
        accz[mt] = __builtin_amdgcn_wmma_f32_16x16x4_f32(
            false, aWs[mt][kc], false, bhs, (short)0, accz[mt], false, false);
      }
    }

    // ---- bias + tanh; park transposed in LDS for re-fragmentation
#pragma unroll
    for (int mt = 0; mt < 2; ++mt) {
      v4f z0, z1;
#pragma unroll
      for (int r = 0; r < 4; ++r) z0[r] = tanhf(accz[mt][r] + bdv[mt][r]);
#pragma unroll
      for (int r = 0; r < 4; ++r) z1[r] = tanhf(accz[mt][r + 4] + bdv[mt][r + 4]);
      float* zp = &zbuf[wv][l15][mt * 16 + hi8];
      *(v4f*)(zp)     = z0;
      *(v4f*)(zp + 4) = z1;
    }

    // ---- r = Wr @ z   (B-frags from LDS)
    v8f accr[2] = {};
#pragma unroll
    for (int kc = 0; kc < 8; ++kc) {
      v2f bz = *(const v2f*)&zbuf[wv][l15][kc * 4 + lh * 2];
#pragma unroll
      for (int mt = 0; mt < 2; ++mt)
        accr[mt] = __builtin_amdgcn_wmma_f32_16x16x4_f32(
            false, aWr[mt][kc], false, bz, (short)0, accr[mt], false, false);
    }

    // ---- h_new = h + r + br ; store f32 chain + f16 H_cat row
#pragma unroll
    for (int mt = 0; mt < 2; ++mt) {
      const int rowb = mt * 16 + hi8;
      const float* hp = hin + (size_t)bt * RC + rowb;
      v4f h0 = *(const v4f*)(hp);
      v4f h1 = *(const v4f*)(hp + 4);
      v4f o0, o1;
#pragma unroll
      for (int r = 0; r < 4; ++r) o0[r] = h0[r] + accr[mt][r] + brv[mt][r];
#pragma unroll
      for (int r = 0; r < 4; ++r) o1[r] = h1[r] + accr[mt][r + 4] + brv[mt][r + 4];
      float* op = hout + (size_t)bt * RC + rowb;
      *(v4f*)(op)     = o0;
      *(v4f*)(op + 4) = o1;
      v8h hh;
#pragma unroll
      for (int r = 0; r < 4; ++r) {
        hh[r]     = (_Float16)o0[r];
        hh[r + 4] = (_Float16)o1[r];
      }
      *(v8h*)(hcat + (size_t)bt * KTOT + layer * DC + rowb) = hh;
    }
  }
}

// ---------------------------------------------------------------------------
// Kernel 3: fused skip GEMM + relu + final 1x1 conv (f16 WMMA 16x16x32)
//   H_cat slabs staged through LDS by the Tensor Data Mover, double-buffered.
// A-frag f16 16x32: halves h: K = (h>=8)*16 + lh*8 + (h&7), M = l15 (+Mtile)
// B-frag f16 32x16: halves h: K = (h>=8)*16 + lh*8 + (h&7), N = l15 (+Ntile)
// ---------------------------------------------------------------------------
__device__ __forceinline__ v16h load_a_f16(const _Float16* wp, int lh) {
  union { v16h v; v8h h[2]; } a;
  a.h[0] = *(const v8h*)(wp + lh * 8);       // K = lh*8 + 0..7
  a.h[1] = *(const v8h*)(wp + lh * 8 + 16);  // K = 16 + lh*8 + 0..7
  return a.v;
}

__global__ __launch_bounds__(256) void k_skip_final(const _Float16* __restrict__ hcat,
                                                    const _Float16* __restrict__ Wsh,
                                                    const float* __restrict__ bs,
                                                    const _Float16* __restrict__ Wfh,
                                                    const float* __restrict__ bf,
                                                    float* __restrict__ out) {
  __shared__ _Float16 hbuf[2][64 * DC];  // TDM-staged H_cat slabs (2 x 4KB)
  __shared__ _Float16 sT[64][264];       // relu(skip) transposed, padded rows
  __shared__ float    bsum[SC];

  const int tid = threadIdx.x;
  {  // sum of skip biases over layers
    float s = 0.0f;
#pragma unroll 4
    for (int i = 0; i < NL; ++i) s += bs[i * SC + tid];
    bsum[tid] = s;
  }

  const int wv   = tid >> 5;
  const int lane = tid & 31;
  const int l15  = lane & 15;
  const int lh   = lane >> 4;
  const int hi8  = lh * 8;
  const int n0   = blockIdx.x * 64;
  const _Float16* hbase = hcat + (size_t)n0 * KTOT;

  // ---- prologue: DMA layers 0 and 1 into the two LDS buffers
  if (wv == 0) {
    tdm_load_tile(hbase + 0 * DC, lds_off(&hbuf[0][0]), DC, 64, KTOT);
    tdm_load_tile(hbase + 1 * DC, lds_off(&hbuf[1][0]), DC, 64, KTOT);
    __builtin_amdgcn_s_wait_tensorcnt(1);  // layer-0 slab resident
  }
  __syncthreads();

  // ---- skip GEMM: K = 1280 in 40 layer-chunks of 32
  v8f acc[2][4] = {};
  for (int i = 0; i < NL; ++i) {
    const _Float16* buf = &hbuf[i & 1][0];
    v16h bfr[4];
#pragma unroll
    for (int nt = 0; nt < 4; ++nt) {
      const _Float16* p = buf + (nt * 16 + l15) * DC + lh * 8;
      union { v16h v; v8h h[2]; } u;
      u.h[0] = *(const v8h*)(p);
      u.h[1] = *(const v8h*)(p + 16);
      bfr[nt] = u.v;
    }
#pragma unroll
    for (int mt = 0; mt < 2; ++mt) {
      const int M = (wv * 2 + mt) * 16 + l15;
      v16h a = load_a_f16(Wsh + ((size_t)i * SC + M) * DC, lh);
#pragma unroll
      for (int nt = 0; nt < 4; ++nt)
        acc[mt][nt] = __builtin_amdgcn_wmma_f32_16x16x32_f16(
            false, a, false, bfr[nt], (short)0, acc[mt][nt], false, false);
    }
    __syncthreads();  // everyone done reading hbuf[i&1]
    if (wv == 0) {
      if (i + 2 < NL) {
        tdm_load_tile(hbase + (size_t)(i + 2) * DC, lds_off(&hbuf[i & 1][0]),
                      DC, 64, KTOT);
        __builtin_amdgcn_s_wait_tensorcnt(1);  // layer i+1 slab resident
      } else {
        __builtin_amdgcn_s_wait_tensorcnt(0);
      }
    }
    __syncthreads();  // hbuf[(i+1)&1] ready for all waves
  }

  // ---- relu(skip + bs_sum) -> LDS transposed f16
#pragma unroll
  for (int mt = 0; mt < 2; ++mt) {
    const int rowb = (wv * 2 + mt) * 16 + hi8;
#pragma unroll
    for (int nt = 0; nt < 4; ++nt) {
      const int col = nt * 16 + l15;
      v8h s8;
#pragma unroll
      for (int r = 0; r < 8; ++r) {
        float v = acc[mt][nt][r] + bsum[rowb + r];
        v = v > 0.0f ? v : 0.0f;
        s8[r] = (_Float16)v;
      }
      *(v8h*)&sT[col][rowb] = s8;
    }
  }
  __syncthreads();

  // ---- final GEMM: out = Wf @ s, K = 256 in 8 chunks of 32
  v8f acco[2][4] = {};
#pragma unroll
  for (int kc = 0; kc < 8; ++kc) {
    v16h bfr[4];
#pragma unroll
    for (int nt = 0; nt < 4; ++nt) {
      const _Float16* p = &sT[nt * 16 + l15][kc * 32 + lh * 8];
      union { v16h v; v8h h[2]; } u;
      u.h[0] = *(const v8h*)(p);
      u.h[1] = *(const v8h*)(p + 16);
      bfr[nt] = u.v;
    }
#pragma unroll
    for (int mt = 0; mt < 2; ++mt) {
      const int M = (wv * 2 + mt) * 16 + l15;
      v16h a = load_a_f16(Wfh + (size_t)M * SC + kc * 32, lh);
#pragma unroll
      for (int nt = 0; nt < 4; ++nt)
        acco[mt][nt] = __builtin_amdgcn_wmma_f32_16x16x32_f16(
            false, a, false, bfr[nt], (short)0, acco[mt][nt], false, false);
    }
  }

  // ---- epilogue: + bf, write [B,256,T]
#pragma unroll
  for (int mt = 0; mt < 2; ++mt) {
    const int rowb = (wv * 2 + mt) * 16 + hi8;
#pragma unroll
    for (int nt = 0; nt < 4; ++nt) {
      const int bt = n0 + nt * 16 + l15;
      const int b = bt >> 15;
      const int t = bt & (TLEN - 1);
      float* op = out + ((size_t)b * OUTC + rowb) * TLEN + t;
#pragma unroll
      for (int r = 0; r < 8; ++r)
        op[(size_t)r * TLEN] = acco[mt][nt][r] + bf[rowb + r];
    }
  }
}

// ---------------------------------------------------------------------------
extern "C" void kernel_launch(void* const* d_in, const int* in_sizes, int n_in,
                              void* d_out, int out_size, void* d_ws, size_t ws_size,
                              hipStream_t stream) {
  const float* x   = (const float*)d_in[0];
  const float* Wst = (const float*)d_in[1];
  const float* bst = (const float*)d_in[2];
  const float* Wd  = (const float*)d_in[3];
  const float* bd  = (const float*)d_in[4];
  const float* Wr  = (const float*)d_in[5];
  const float* br  = (const float*)d_in[6];
  const float* Ws  = (const float*)d_in[7];
  const float* bs  = (const float*)d_in[8];
  const float* Wf  = (const float*)d_in[9];
  const float* bf  = (const float*)d_in[10];
  float* out = (float*)d_out;

  // workspace: h ping (f32) | h pong (f32) | H_cat (f16) | Ws f16 | Wf f16
  float* hA = (float*)d_ws;
  float* hB = hA + (size_t)BT * RC;
  _Float16* hcat = (_Float16*)(hB + (size_t)BT * RC);
  _Float16* Wsh  = hcat + (size_t)BT * KTOT;
  _Float16* Wfh  = Wsh + (size_t)NL * SC * DC;

  const int nWs = NL * SC * DC;   // 327680
  const int nWf = OUTC * SC;      // 65536
  k_cvt<<<(nWs + 255) / 256, 256, 0, stream>>>(Ws, Wsh, nWs);
  k_cvt<<<(nWf + 255) / 256, 256, 0, stream>>>(Wf, Wfh, nWf);

  k_start<<<BT / 256, 256, 0, stream>>>(x, Wst, bst, hA);

  const float* hin = hA;
  float* hout = hB;
  for (int i = 0; i < NL; ++i) {
    const int d = 1 << (i % LAYERS);
    k_layer<<<BT / 512, 256, 0, stream>>>(hin, hout, hcat, Wd, bd, Wr, br, i, d);
    float* tmp = hout;
    hout = (float*)hin;
    hin = tmp;
  }

  k_skip_final<<<BT / 64, 256, 0, stream>>>(hcat, Wsh, bs, Wfh, bf, out);
}
